// WindowMultipleSelfAttention_54236847014192
// MI455X (gfx1250) — matrix operand order repaired and checked by reading, not compiled, not run on hardware
//
#include <hip/hip_runtime.h>
#include <hip/hip_bf16.h>

// Swin window attention for MI455X (gfx1250), wave32 + v_wmma_f32_16x16x32_f16.
// B=4096 windows, N=49 tokens (padded to 64), C=128, H=4 heads, hd=32.

typedef __attribute__((ext_vector_type(16))) _Float16 v16h;
typedef __attribute__((ext_vector_type(8)))  float    v8f;
typedef __attribute__((ext_vector_type(4)))  unsigned int u32x4;
typedef __attribute__((ext_vector_type(4)))  float    f32x4;

#define NTOK 49
#define NPAD 64
#define CDIM 128
#define NH 4
#define NWIN 64
#define BATCH 4096

// padded LDS strides (bank-conflict-free: stride*lane spreads across 64 banks)
#define XS 136   // f16 stride for 128-wide arrays (xA, q, k, ctx)  -> 68 words
#define PS 72    // f16 stride for 64-wide arrays (attnP, vT)       -> 36 words
#define TS 68    // f32 stride for transposed score matrix          -> 68 words

// workspace layout (bytes):
//   [0)        qkv_wT : f16[384*128]   -- qkv_w transposed, f16
//   [98304)    proj_wT: f16[128*128]   -- proj_w transposed, f16
//   [131072)   biasH  : f32[4*49*49]   -- gathered rel-pos bias
#define WS_PROJ_OFF 98304
#define WS_BIAS_OFF 131072

// LDS regions (byte offsets), all 16B aligned:
//   R1 @0      : xA f16[64][XS] -> attnT f32[4][64][TS] -> ctx f16[64][XS]   (69632)
//   R2 @69632  : q f16[64][XS] @+0, k f16[64][XS] @+17408  -> attnP f16[4][64][PS] (36864)
//   R3 @106496 : vT f16[128][PS]                                            (18432)
//   R4 @124928 : maskS f32[49*49]                                           (9604)
#define SM_R2    69632
#define SM_K     (SM_R2 + 17408)
#define SM_VT    106496
#define SM_MASK  124928
#define SM_TOTAL 134544

__device__ __forceinline__ v16h ldfrag2(const _Float16* p0, const _Float16* p1) {
    union { u32x4 u[2]; v16h h; } f;
    f.u[0] = *(const u32x4*)p0;   // 8 f16 (16B)
    f.u[1] = *(const u32x4*)p1;   // 8 f16 (16B)
    return f.h;
}
__device__ __forceinline__ v16h ldfragc(const _Float16* p) { return ldfrag2(p, p + 16); }

__device__ __forceinline__ v8f wmma_f16(v16h a, v16h b, v8f c) {
    return __builtin_amdgcn_wmma_f32_16x16x32_f16(false, a, false, b, (short)0, c, false, false);
}

// ---------------- prep: transpose/convert weights, gather bias -----------------
__global__ __launch_bounds__(256) void prep_kernel(const float* __restrict__ qkv_w,
                                                   const float* __restrict__ proj_w,
                                                   const float* __restrict__ bias_table,
                                                   unsigned char* __restrict__ ws) {
    _Float16* qkv_wT  = (_Float16*)(ws);
    _Float16* proj_wT = (_Float16*)(ws + WS_PROJ_OFF);
    float*    biasH   = (float*)(ws + WS_BIAS_OFF);
    int tid = blockIdx.x * 256 + threadIdx.x;
    if (tid < 384 * 128) {                       // qkv_wT[n][k] = qkv_w[k][n]
        int n = tid >> 7, k = tid & 127;
        qkv_wT[tid] = (_Float16)qkv_w[k * 384 + n];
    }
    int t2 = tid - 384 * 128;
    if (t2 >= 0 && t2 < 128 * 128) {             // proj_wT[n][k] = proj_w[k][n]
        int n = t2 >> 7, k = t2 & 127;
        proj_wT[t2] = (_Float16)proj_w[k * 128 + n];
    }
    int t3 = tid - (384 * 128 + 128 * 128);
    if (t3 >= 0 && t3 < NH * NTOK * NTOK) {      // biasH[h][n][m]
        int h = t3 / (NTOK * NTOK);
        int r = t3 % (NTOK * NTOK);
        int n = r / NTOK, m = r % NTOK;
        int idx = (n / 7 - m / 7 + 6) * 13 + (n % 7 - m % 7 + 6);
        biasH[t3] = bias_table[idx * NH + h];
    }
}

// ---------------- main: one workgroup (8 wave32) per window --------------------
__global__ __launch_bounds__(256) void swin_attn_kernel(const float* __restrict__ x,
                                                        const float* __restrict__ mask,
                                                        const float* __restrict__ qkv_b,
                                                        const float* __restrict__ proj_b,
                                                        const unsigned char* __restrict__ ws,
                                                        float* __restrict__ out) {
    extern __shared__ __align__(16) unsigned char smem[];
    _Float16* xA    = (_Float16*)(smem);               // [64][XS]
    float*    attnT = (float*)(smem);                  // [4][64 m'][TS] (transposed scores)
    _Float16* ctx   = (_Float16*)(smem);               // [64][XS]
    _Float16* qs    = (_Float16*)(smem + SM_R2);       // [64][XS]
    _Float16* ks    = (_Float16*)(smem + SM_K);        // [64][XS]
    _Float16* attnP = (_Float16*)(smem + SM_R2);       // [4][64][PS]
    _Float16* vT    = (_Float16*)(smem + SM_VT);       // [128][PS]
    float*    maskS = (float*)(smem + SM_MASK);        // [49*49]

    const _Float16* qkv_wT  = (const _Float16*)ws;
    const _Float16* proj_wT = (const _Float16*)(ws + WS_PROJ_OFF);
    const float*    biasH   = (const float*)(ws + WS_BIAS_OFF);

    const int b    = blockIdx.x;
    const int tid  = threadIdx.x;
    const int wvs  = __builtin_amdgcn_readfirstlane(tid >> 5);  // scalar wave id 0..7
    const int lane = tid & 31;
    const int lj   = lane & 15;
    const int half = lane >> 4;
    const float scale = 0.08838834764831845f;   // 128^-0.5 (embed_dim, per reference)

    // ---- stage 0: x -> f16 LDS (vectorized), pad rows 49..63 zero; stage mask ----
    {
        const float* xb = x + (size_t)b * (NTOK * CDIM);
        for (int i = tid; i < NPAD * CDIM / 4; i += 256) {
            int row = i >> 5;
            int c4  = (i & 31) << 2;
            f32x4 v = {};
            if (row < NTOK) v = *(const f32x4*)(xb + row * CDIM + c4);
            union { _Float16 h[4]; unsigned long long u; } p;
            p.h[0] = (_Float16)v[0]; p.h[1] = (_Float16)v[1];
            p.h[2] = (_Float16)v[2]; p.h[3] = (_Float16)v[3];
            *(unsigned long long*)&xA[row * XS + c4] = p.u;
        }
        int wm = b & (NWIN - 1);
        const float* mrow = mask + (size_t)wm * (NTOK * NTOK);
        for (int i = tid; i < NTOK * NTOK; i += 256) maskS[i] = mrow[i];
    }
    __syncthreads();

    // ---- stage 1: qkv = xA @ qkv_w + b  (mt fixed per wave; A frags hoisted) ----
    {
        const int mt  = wvs & 3;
        const int nt0 = wvs >> 2;
        const _Float16* ar = &xA[(mt * 16 + lj) * XS + half * 8];
        v16h a0 = ldfrag2(ar +  0, ar + 16);
        v16h a1 = ldfrag2(ar + 32, ar + 48);
        v16h a2 = ldfrag2(ar + 64, ar + 80);
        v16h a3 = ldfrag2(ar + 96, ar + 112);
        const int rbase = mt * 16 + half * 8;
        for (int nt = nt0; nt < 24; nt += 2) {
            const _Float16* bp = &qkv_wT[(nt * 16 + lj) * 128 + half * 8];
            v8f acc = {};
            acc = wmma_f16(a0, ldfragc(bp),      acc);
            acc = wmma_f16(a1, ldfragc(bp + 32), acc);
            acc = wmma_f16(a2, ldfragc(bp + 64), acc);
            acc = wmma_f16(a3, ldfragc(bp + 96), acc);
            int col  = nt * 16 + lj;
            float bias = qkv_b[col];
            int c    = col & 127;
            int sec  = nt >> 3;                  // scalar: 0=q 1=k 2=v
            if (sec == 0) {
                #pragma unroll
                for (int r = 0; r < 8; ++r)
                    qs[(rbase + r) * XS + c] = (_Float16)((acc[r] + bias) * scale);
            } else if (sec == 1) {
                #pragma unroll
                for (int r = 0; r < 8; ++r)
                    ks[(rbase + r) * XS + c] = (_Float16)(acc[r] + bias);
            } else {                             // v transposed, contiguous -> 1x b128
                union { _Float16 h[8]; u32x4 u; } pk;
                #pragma unroll
                for (int r = 0; r < 8; ++r) pk.h[r] = (_Float16)(acc[r] + bias);
                *(u32x4*)&vT[c * PS + rbase] = pk.u;
            }
        }
    }
    __syncthreads();

    // ---- stage 2: scores = q @ k^T per head, store TRANSPOSED [m'][query] ----
    {
        const int nt = wvs & 3;                  // key tile fixed per wave
        const int u0 = wvs >> 2;
        const int colN = nt * 16 + lj;           // key index m'
        for (int i = 0; i < 8; ++i) {
            int u = u0 + 2 * i;
            int h = u >> 2, mt = u & 3;
            const _Float16* ap = &qs[(mt * 16 + lj) * XS + h * 32 + half * 8];
            const _Float16* bp = &ks[colN * XS + h * 32 + half * 8];
            v8f acc = {};
            acc = wmma_f16(ldfrag2(ap, ap + 16), ldfragc(bp), acc);
            int rbase = mt * 16 + half * 8;
            union { v8f v; u32x4 u4[2]; } pk; pk.v = acc;
            float* dst = &attnT[(h * 64 + colN) * TS + rbase];  // contiguous rows
            *(u32x4*)dst       = pk.u4[0];
            *(u32x4*)(dst + 4) = pk.u4[1];
        }
    }
    __syncthreads();

    // ---- stage 3: softmax rows (fuse rel-pos bias + shift mask), write P f16 ----
    if (tid < NH * NTOK) {
        int h = tid / NTOK, n = tid % NTOK;      // query row n of head h
        float* colS = &attnT[h * 64 * TS + n];   // element m' at colS[m'*TS]
        const float* brow = &biasH[(h * NTOK + n) * NTOK];
        const float* mrow = &maskS[n * NTOK];
        float mx = -3.0e38f;
        for (int m = 0; m < NTOK; ++m) {
            float v = colS[m * TS] + brow[m] + mrow[m];
            colS[m * TS] = v;
            mx = fmaxf(mx, v);
        }
        _Float16* prow = &attnP[(h * 64 + n) * PS];
        float sum = 0.0f;
        for (int m = 0; m < NTOK; ++m) {
            float e = __expf(colS[m * TS] - mx);
            sum += e;
            prow[m] = (_Float16)e;
        }
        float inv = 1.0f / sum;
        for (int m = 0; m < NTOK; ++m)
            prow[m] = (_Float16)((float)prow[m] * inv);
        for (int m = NTOK; m < 64; ++m)          // zero pad cols -> padded k/v inert
            prow[m] = (_Float16)0.0f;
    }
    __syncthreads();

    // ---- stage 4: ctx = P @ v per head (K=64 in 2 steps) ----
    {
        const int mt = (wvs >> 1) & 3;
        const int nt = wvs & 1;
        const int rbase = mt * 16 + half * 8;
        for (int h = 0; h < NH; ++h) {
            v8f acc = {};
            #pragma unroll
            for (int kb = 0; kb < 64; kb += 32) {
                const _Float16* ap = &attnP[(h * 64 + mt * 16 + lj) * PS + kb + half * 8];
                const _Float16* bp = &vT[(h * 32 + nt * 16 + lj) * PS + kb + half * 8];
                acc = wmma_f16(ldfrag2(ap, ap + 16), ldfragc(bp), acc);
            }
            int col = h * 32 + nt * 16 + lj;
            #pragma unroll
            for (int r = 0; r < 8; ++r)
                ctx[(rbase + r) * XS + col] = (_Float16)acc[r];
        }
    }
    __syncthreads();

    // ---- stage 5: out = ctx @ proj_w + proj_b (mt fixed per wave; A hoisted) ----
    {
        const int mt  = wvs & 3;
        const int nt0 = wvs >> 2;
        const _Float16* ar = &ctx[(mt * 16 + lj) * XS + half * 8];
        v16h a0 = ldfrag2(ar +  0, ar + 16);
        v16h a1 = ldfrag2(ar + 32, ar + 48);
        v16h a2 = ldfrag2(ar + 64, ar + 80);
        v16h a3 = ldfrag2(ar + 96, ar + 112);
        const int rbase = mt * 16 + half * 8;
        float* ob = out + (size_t)b * (NTOK * CDIM);
        for (int nt = nt0; nt < 8; nt += 2) {
            const _Float16* bp = &proj_wT[(nt * 16 + lj) * 128 + half * 8];
            v8f acc = {};
            acc = wmma_f16(a0, ldfragc(bp),      acc);
            acc = wmma_f16(a1, ldfragc(bp + 32), acc);
            acc = wmma_f16(a2, ldfragc(bp + 64), acc);
            acc = wmma_f16(a3, ldfragc(bp + 96), acc);
            int col = nt * 16 + lj;
            float pb = proj_b[col];
            if (mt < 3) {                        // scalar branch: all 8 rows valid
                #pragma unroll
                for (int r = 0; r < 8; ++r)
                    ob[(rbase + r) * CDIM + col] = acc[r] + pb;
            } else {                             // mt==3: only row 48 (half==0, r==0)
                if (half == 0) ob[48 * CDIM + col] = acc[0] + pb;
            }
        }
    }
}

extern "C" void kernel_launch(void* const* d_in, const int* in_sizes, int n_in,
                              void* d_out, int out_size, void* d_ws, size_t ws_size,
                              hipStream_t stream) {
    const float* x          = (const float*)d_in[0];   // [4096,49,128]
    const float* mask       = (const float*)d_in[1];   // [64,49,49]
    const float* qkv_w      = (const float*)d_in[2];   // [128,384]
    const float* qkv_b      = (const float*)d_in[3];   // [384]
    const float* proj_w     = (const float*)d_in[4];   // [128,128]
    const float* proj_b     = (const float*)d_in[5];   // [128]
    const float* bias_table = (const float*)d_in[6];   // [169,4]
    unsigned char* ws = (unsigned char*)d_ws;          // needs >= ~170 KB

    // prep: 49152 + 16384 + 9604 = 75140 work items
    prep_kernel<<<(75140 + 255) / 256, 256, 0, stream>>>(qkv_w, proj_w, bias_table, ws);

    // main: one workgroup per window, ~131 KB dynamic LDS (2 WGs per 320KB WGP)
    swin_attn_kernel<<<BATCH, 256, SM_TOTAL, stream>>>(x, mask, qkv_b, proj_b, ws,
                                                       (float*)d_out);
    (void)in_sizes; (void)n_in; (void)out_size; (void)ws_size;
}
